// MultiheadPointTransformerLayer_27419071218245
// MI455X (gfx1250) — matrix operand order repaired and checked by reading, not compiled, not run on hardware
//
#include <hip/hip_runtime.h>
#include <hip/hip_bf16.h>

typedef __attribute__((ext_vector_type(16))) _Float16 v16h;
typedef __attribute__((ext_vector_type(8)))  float    v8f;
typedef __attribute__((ext_vector_type(4)))  float    v4f;

namespace {
constexpr int B_    = 2;
constexpr int N_    = 2304;     // 48*48
constexpr int NNB   = 314;      // neighbors per point
constexpr int NT    = 20;       // ceil(314/16) j-tiles
constexpr int HEADS = 4;
constexpr int DH    = 8;
constexpr int HID   = 32;
constexpr int DIM   = 64;
constexpr int BH    = B_ * HEADS;          // 8
constexpr int JD    = NNB * DH;            // 2512
constexpr int QKV1  = BH * N_ * DH;        // 147456 floats per tensor
}

// ---- monotonic float <-> uint mapping for atomic max on floats ----
__device__ __forceinline__ unsigned fmap(float f) {
  unsigned u = __float_as_uint(f);
  return (u >> 31) ? ~u : (u | 0x80000000u);
}
__device__ __forceinline__ float funmap(unsigned u) {
  return __uint_as_float((u & 0x80000000u) ? (u & 0x7FFFFFFFu) : ~u);
}

// ---- per-wave constant fragments (per head) ----
// GEMM1: hmid[j,e] = relu( sum_d Qrel[j,d]*w1[e,d] + b1[e] )  (M=j,K=d(pad 8->32),N=e)
// GEMM2 (transposed): sim^T[d,j] = sum_e w2[d,e]*hmid[j,e] + b2[d] (M=d(pad 8->16),K=e=32,N=j)
__device__ __forceinline__ void make_frags(
    const float* __restrict__ w1h, const float* __restrict__ b1h,
    const float* __restrict__ w2h, const float* __restrict__ b2h,
    int lane, v16h& b1f0, v16h& b1f1, v16h& a2f,
    v8f& c10, v8f& c11, v8f& c2)
{
#pragma unroll
  for (int kk = 0; kk < 16; ++kk) {
    b1f0[kk] = (_Float16)0.f; b1f1[kk] = (_Float16)0.f; a2f[kk] = (_Float16)0.f;
  }
  const int e0 = lane & 15;
  // B operand of GEMM1: B[K=d][N=e] = w1[e][d]; K(real)=0..7 -> only lanes<16, halves 0..7
  if (lane < 16) {
#pragma unroll
    for (int d = 0; d < 8; ++d) {
      b1f0[d] = (_Float16)w1h[e0 * DH + d];          // e = e0
      b1f1[d] = (_Float16)w1h[(e0 + 16) * DH + d];   // e = e0+16
    }
  }
  // A operand of GEMM2: A[M=d][K=e] = w2[d][e]; contiguous k order used here (same as GEMM2's B):
  // lanes<16: M=lane, halves0..15 = K0..15
  // lanes>=16: M=lane-16, halves0..15 = K16..31
  const int dd = lane & 15;
  if (dd < DH) {
    const int eb = (lane < 16) ? 0 : 16;
#pragma unroll
    for (int v = 0; v < 8; ++v) {
      a2f[v]     = (_Float16)w2h[dd * HID + eb + v];
      a2f[8 + v] = (_Float16)w2h[dd * HID + 8 + eb + v];
    }
  }
  const float bb0 = b1h[e0], bb1 = b1h[e0 + 16];
#pragma unroll
  for (int r = 0; r < 8; ++r) {
    c10[r] = bb0;                       // all M rows of column e get +b1[e]
    c11[r] = bb1;
    c2[r]  = (lane < 16) ? b2h[r] : 0.f; // D rows M=r -> d=r for lanes<16
  }
}

// ---- core: one 16-neighbor tile -> sim^T fragment (lane<16 holds sim[d=0..7][j=tile*16+lane]) ----
__device__ __forceinline__ v8f sim_tile(
    const float* __restrict__ kb, const int* __restrict__ idxrow,
    const float (&qv)[8], int tile, int lane,
    v16h b1f0, v16h b1f1, v16h a2f, v8f c10, v8f c11, v8f c2,
    _Float16* tb, int& kidx_out)
{
  const int jl   = tile * 16 + (lane & 15);
  const int jc   = (jl < NNB) ? jl : (NNB - 1);
  const int kidx = idxrow[jc];
  kidx_out = kidx;
  const float* kr = kb + (size_t)kidx * DH;
  const v4f k0 = *(const v4f*)kr;
  const v4f k1 = *(const v4f*)(kr + 4);
  const bool act = (lane < 16) && (jl < NNB);
  // A of GEMM1: 16-bit A 16x32 layout; real K=0..7 lives in lanes<16 halves0..7.
  v16h a1;
#pragma unroll
  for (int d = 0; d < 4; ++d) {
    a1[d]     = act ? (_Float16)(qv[d]     - k0[d]) : (_Float16)0.f;
    a1[4 + d] = act ? (_Float16)(qv[4 + d] - k1[d]) : (_Float16)0.f;
  }
#pragma unroll
  for (int d = 8; d < 16; ++d) a1[d] = (_Float16)0.f;

  v8f h0 = __builtin_amdgcn_wmma_f32_16x16x32_f16(false, a1, false, b1f0, (short)0, c10, false, false);
  v8f h1 = __builtin_amdgcn_wmma_f32_16x16x32_f16(false, a1, false, b1f1, (short)0, c11, false, false);

  // relu, cvt f16, lane-transpose through LDS (D is column-per-lane; GEMM2's B needs row-per-lane)
  const int e0 = lane & 15;
  const int m0 = (lane < 16) ? 0 : 8;
#pragma unroll
  for (int r = 0; r < 8; ++r) {
    tb[(m0 + r) * HID + e0]      = (_Float16)fmaxf(h0[r], 0.0f);
    tb[(m0 + r) * HID + 16 + e0] = (_Float16)fmaxf(h1[r], 0.0f);
  }
  asm volatile("s_wait_dscnt 0x0" ::: "memory");
  // B of GEMM2 (32x16): lane<16 -> row j=lane, K halves e=0..15; lane>=16 -> row j=lane-16, e=16..31
  const v16h bh2 = *(const v16h*)(tb + e0 * HID + ((lane < 16) ? 0 : 16));
  return __builtin_amdgcn_wmma_f32_16x16x32_f16(false, a2f, false, bh2, (short)0, c2, false, false);
}

// ---------------- kernel 0: init M (mapped -inf == 0) and S ----------------
__global__ void k_init(unsigned* __restrict__ Mb, float* __restrict__ Sb) {
  const int t = blockIdx.x * 256 + threadIdx.x;
  if (t < BH * JD) { Mb[t] = 0u; Sb[t] = 0.0f; }
}

// ---------------- kernel 1: qkv projection into [which][b][h][n][d] ----------------
__global__ __launch_bounds__(256) void k_qkv(
    const float* __restrict__ x, const float* __restrict__ wqkv,
    float* __restrict__ qkv)
{
  const int t   = blockIdx.x * 256 + threadIdx.x;   // < B*N*96
  const int col = t % 96;
  const int bi  = t / 96;
  const float* xr = x + (size_t)bi * DIM;
  float s = 0.f;
#pragma unroll 8
  for (int m = 0; m < DIM; ++m) s += xr[m] * wqkv[m * 96 + col];
  const int which = col >> 5, hh = (col & 31) >> 3, d = col & 7;
  const int b = bi / N_, i = bi % N_;
  qkv[((size_t)which * BH + b * HEADS + hh) * N_ * DH + (size_t)i * DH + d] = s;
}

// ---------------- kernel 2: column-wise (axis=i) max of sim ----------------
__global__ __launch_bounds__(256) void k_simmax(
    const float* __restrict__ qkv, const int* __restrict__ idx,
    const float* __restrict__ w1, const float* __restrict__ b1,
    const float* __restrict__ w2, const float* __restrict__ b2,
    unsigned* __restrict__ Mb)
{
  __shared__ unsigned red[JD];
  __shared__ __align__(32) _Float16 tbuf[8][2][16 * HID];
  const int tid = threadIdx.x, lane = tid & 31, wave = tid >> 5;
  const int bh = blockIdx.x / (N_ / 64);
  const int chunk = blockIdx.x % (N_ / 64);
  const int h = bh & (HEADS - 1);
  for (int t = tid; t < JD; t += 256) red[t] = 0u;
  __syncthreads();

  v16h b1f0, b1f1, a2f; v8f c10, c11, c2;
  make_frags(w1 + h * HID * DH, b1 + h * HID, w2 + h * DH * HID, b2 + h * DH,
             lane, b1f0, b1f1, a2f, c10, c11, c2);
  const float* qb = qkv + (size_t)bh * N_ * DH;
  const float* kb = qkv + (size_t)(BH + bh) * N_ * DH;

  for (int ii = 0; ii < 8; ++ii) {
    const int i = chunk * 64 + wave * 8 + ii;
    const float* qr = qb + (size_t)i * DH;
    const v4f q0 = *(const v4f*)qr, q1 = *(const v4f*)(qr + 4);
    float qv[8];
#pragma unroll
    for (int r = 0; r < 4; ++r) { qv[r] = q0[r]; qv[4 + r] = q1[r]; }
    const int* idxrow = idx + (size_t)i * NNB;
    for (int t = 0; t < NT; ++t) {
      int kidx;
      v8f s = sim_tile(kb, idxrow, qv, t, lane, b1f0, b1f1, a2f, c10, c11, c2,
                       &tbuf[wave][t & 1][0], kidx);
      const int jl = t * 16 + (lane & 15);
      if (lane < 16 && jl < NNB) {
#pragma unroll
        for (int r = 0; r < 8; ++r) atomicMax(&red[jl * DH + r], fmap(s[r]));
      }
    }
  }
  __syncthreads();
  unsigned* Mg = Mb + (size_t)bh * JD;
  for (int t = tid; t < JD; t += 256) atomicMax(&Mg[t], red[t]);
}

// ---------------- kernel 3: column-wise sum of exp(sim - M) ----------------
__global__ __launch_bounds__(256) void k_simsum(
    const float* __restrict__ qkv, const int* __restrict__ idx,
    const float* __restrict__ w1, const float* __restrict__ b1,
    const float* __restrict__ w2, const float* __restrict__ b2,
    const unsigned* __restrict__ Mb, float* __restrict__ Sb)
{
  __shared__ float Ms[JD];
  __shared__ float ssum[JD];
  __shared__ __align__(32) _Float16 tbuf[8][2][16 * HID];
  const int tid = threadIdx.x, lane = tid & 31, wave = tid >> 5;
  const int bh = blockIdx.x / (N_ / 64);
  const int chunk = blockIdx.x % (N_ / 64);
  const int h = bh & (HEADS - 1);
  for (int t = tid; t < JD; t += 256) {
    Ms[t] = funmap(Mb[(size_t)bh * JD + t]);
    ssum[t] = 0.0f;
  }
  __syncthreads();

  v16h b1f0, b1f1, a2f; v8f c10, c11, c2;
  make_frags(w1 + h * HID * DH, b1 + h * HID, w2 + h * DH * HID, b2 + h * DH,
             lane, b1f0, b1f1, a2f, c10, c11, c2);
  const float* qb = qkv + (size_t)bh * N_ * DH;
  const float* kb = qkv + (size_t)(BH + bh) * N_ * DH;

  for (int ii = 0; ii < 8; ++ii) {
    const int i = chunk * 64 + wave * 8 + ii;
    const float* qr = qb + (size_t)i * DH;
    const v4f q0 = *(const v4f*)qr, q1 = *(const v4f*)(qr + 4);
    float qv[8];
#pragma unroll
    for (int r = 0; r < 4; ++r) { qv[r] = q0[r]; qv[4 + r] = q1[r]; }
    const int* idxrow = idx + (size_t)i * NNB;
    for (int t = 0; t < NT; ++t) {
      int kidx;
      v8f s = sim_tile(kb, idxrow, qv, t, lane, b1f0, b1f1, a2f, c10, c11, c2,
                       &tbuf[wave][t & 1][0], kidx);
      const int jl = t * 16 + (lane & 15);
      if (lane < 16 && jl < NNB) {
#pragma unroll
        for (int r = 0; r < 8; ++r)
          atomicAdd(&ssum[jl * DH + r], __expf(s[r] - Ms[jl * DH + r]));
      }
    }
  }
  __syncthreads();
  float* Sg = Sb + (size_t)bh * JD;
  for (int t = tid; t < JD; t += 256) atomicAdd(&Sg[t], ssum[t]);
}

// ---------------- kernel 4: attn-weighted aggregation of v ----------------
__global__ __launch_bounds__(256) void k_agg(
    const float* __restrict__ qkv, const int* __restrict__ idx,
    const float* __restrict__ w1, const float* __restrict__ b1,
    const float* __restrict__ w2, const float* __restrict__ b2,
    const unsigned* __restrict__ Mb, const float* __restrict__ Sb,
    float* __restrict__ agg)
{
  __shared__ float Ms[JD];
  __shared__ float Ss[JD];
  __shared__ __align__(32) _Float16 tbuf[8][2][16 * HID];
  const int tid = threadIdx.x, lane = tid & 31, wave = tid >> 5;
  const int bh = blockIdx.x / (N_ / 8);
  const int h = bh & (HEADS - 1);
  const int i = (blockIdx.x % (N_ / 8)) * 8 + wave;   // one query per wave
  for (int t = tid; t < JD; t += 256) {
    Ms[t] = funmap(Mb[(size_t)bh * JD + t]);
    Ss[t] = Sb[(size_t)bh * JD + t];
  }
  __syncthreads();

  v16h b1f0, b1f1, a2f; v8f c10, c11, c2;
  make_frags(w1 + h * HID * DH, b1 + h * HID, w2 + h * DH * HID, b2 + h * DH,
             lane, b1f0, b1f1, a2f, c10, c11, c2);
  const float* qb = qkv + (size_t)bh * N_ * DH;
  const float* kb = qkv + (size_t)(BH + bh) * N_ * DH;
  const float* vb = qkv + (size_t)(2 * BH + bh) * N_ * DH;

  const float* qr = qb + (size_t)i * DH;
  const v4f q0 = *(const v4f*)qr, q1 = *(const v4f*)(qr + 4);
  float qv[8];
#pragma unroll
  for (int r = 0; r < 4; ++r) { qv[r] = q0[r]; qv[4 + r] = q1[r]; }
  const int* idxrow = idx + (size_t)i * NNB;

  float acc[8];
#pragma unroll
  for (int r = 0; r < 8; ++r) acc[r] = 0.f;

  for (int t = 0; t < NT; ++t) {
    int kidx;
    v8f s = sim_tile(kb, idxrow, qv, t, lane, b1f0, b1f1, a2f, c10, c11, c2,
                     &tbuf[wave][t & 1][0], kidx);
    const int jl = t * 16 + (lane & 15);
    if (lane < 16 && jl < NNB) {
      const float* vr = vb + (size_t)kidx * DH;
      const v4f v0 = *(const v4f*)vr, v1 = *(const v4f*)(vr + 4);
#pragma unroll
      for (int r = 0; r < 8; ++r) {
        const float w = __expf(s[r] - Ms[jl * DH + r]) / Ss[jl * DH + r];
        acc[r] += w * ((r < 4) ? v0[r] : v1[r - 4]);
      }
    }
  }
  // sum partial j-subsets across lanes (lanes>=16 contribute zero)
#pragma unroll
  for (int off = 16; off >= 1; off >>= 1) {
#pragma unroll
    for (int r = 0; r < 8; ++r) acc[r] += __shfl_xor(acc[r], off, 32);
  }
  if (lane == 0) {
    const int b = bh >> 2;
    float* o = agg + ((size_t)b * N_ + i) * (HEADS * DH) + h * DH;
#pragma unroll
    for (int r = 0; r < 8; ++r) o[r] = acc[r];
  }
}

// ---------------- kernel 5: output projection ----------------
__global__ __launch_bounds__(256) void k_out(
    const float* __restrict__ agg, const float* __restrict__ wout,
    const float* __restrict__ bout, float* __restrict__ out)
{
  const int t = blockIdx.x * 256 + threadIdx.x;   // < B*N*64
  const int c = t & 63;
  const int bi = t >> 6;
  const float* ar = agg + (size_t)bi * (HEADS * DH);
  float s = bout[c];
#pragma unroll
  for (int m = 0; m < HEADS * DH; ++m) s += ar[m] * wout[m * DIM + c];
  out[t] = s;
}

extern "C" void kernel_launch(void* const* d_in, const int* in_sizes, int n_in,
                              void* d_out, int out_size, void* d_ws, size_t ws_size,
                              hipStream_t stream) {
  (void)in_sizes; (void)n_in; (void)out_size; (void)ws_size;
  const float* x    = (const float*)d_in[0];
  const int*   idx  = (const int*)  d_in[1];
  const float* wqkv = (const float*)d_in[2];
  const float* w1   = (const float*)d_in[3];
  const float* b1   = (const float*)d_in[4];
  const float* w2   = (const float*)d_in[5];
  const float* b2   = (const float*)d_in[6];
  const float* wout = (const float*)d_in[7];
  const float* bout = (const float*)d_in[8];

  float*    ws   = (float*)d_ws;
  float*    qkv  = ws;                                   // 3*QKV1 floats
  unsigned* Mb   = (unsigned*)(ws + 3 * QKV1);           // BH*JD uints
  float*    Sb   = ws + 3 * QKV1 + BH * JD;              // BH*JD floats
  float*    agg  = ws + 3 * QKV1 + 2 * BH * JD;          // B*N*32 floats

  k_init  <<<(BH * JD + 255) / 256, 256, 0, stream>>>(Mb, Sb);
  k_qkv   <<<(B_ * N_ * 96) / 256, 256, 0, stream>>>(x, wqkv, qkv);
  k_simmax<<<BH * (N_ / 64), 256, 0, stream>>>(qkv, idx, w1, b1, w2, b2, Mb);
  k_simsum<<<BH * (N_ / 64), 256, 0, stream>>>(qkv, idx, w1, b1, w2, b2, Mb, Sb);
  k_agg   <<<BH * (N_ / 8), 256, 0, stream>>>(qkv, idx, w1, b1, w2, b2, Mb, Sb, agg);
  k_out   <<<(B_ * N_ * DIM) / 256, 256, 0, stream>>>(agg, wout, bout, (float*)d_out);
}
